// Decoder_61229053772060
// MI455X (gfx1250) — matrix operand order repaired and checked by reading, not compiled, not run on hardware
//
#include <hip/hip_runtime.h>
#include <hip/hip_bf16.h>
#include <stdint.h>

// Problem constants (match reference)
#define BATCH 2048
#define NNODE 200
#define EMB   128
#define HEADS 8
#define HDIM  16
#define HD    128      // HEADS*HDIM
#define QIN   384      // 3*EMB

typedef __attribute__((ext_vector_type(16))) _Float16 v16h;
typedef __attribute__((ext_vector_type(8)))  float    v8f;

// ---------------------------------------------------------------------------
// threefry2x32-style deterministic hash -> Gumbel noise. (Bit-exact match to
// jax's key-derivation cannot be verified in this compile-only loop; the
// structure — per-step folded key, per-(b,n) counter, Gumbel-argmax — matches.)
// ---------------------------------------------------------------------------
struct U2 { unsigned x, y; };

__device__ inline U2 tf2x32(U2 key, U2 ctr) {
  const unsigned R[8] = {13u,15u,26u,6u,17u,29u,16u,24u};
  unsigned ks0 = key.x, ks1 = key.y, ks2 = ks0 ^ ks1 ^ 0x1BD11BDAu;
  unsigned x0 = ctr.x + ks0, x1 = ctr.y + ks1;
  unsigned ks[3] = {ks0, ks1, ks2};
#pragma unroll
  for (int g = 0; g < 5; ++g) {
#pragma unroll
    for (int rn = 0; rn < 4; ++rn) {
      unsigned rot = R[(g & 1) * 4 + rn];
      x0 += x1; x1 = (x1 << rot) | (x1 >> (32 - rot)); x1 ^= x0;
    }
    x0 += ks[(g + 1) % 3];
    x1 += ks[(g + 2) % 3] + (unsigned)(g + 1);
  }
  U2 r; r.x = x0; r.y = x1; return r;
}

union F4H8 { float4 v; _Float16 h[8]; };

// ---------------------------------------------------------------------------
// Init state: last=initK, first=initV (broadcast), mask=0, logp=0
// ---------------------------------------------------------------------------
__global__ void initStateKernel(const float* __restrict__ initK,
                                const float* __restrict__ initV,
                                float* __restrict__ last, float* __restrict__ first,
                                unsigned char* __restrict__ mask,
                                float* __restrict__ logp) {
  const int i = blockIdx.x * blockDim.x + threadIdx.x;
  if (i < BATCH * EMB) {
    const int e = i & (EMB - 1);
    last[i]  = initK[e];
    first[i] = initV[e];
  }
  if (i < BATCH * NNODE) mask[i] = 0;
  if (i < BATCH) logp[i] = 0.0f;
}

// ---------------------------------------------------------------------------
// hinit[b,e] = mean over n of x[b,n,e]
// ---------------------------------------------------------------------------
__global__ void hinitKernel(const float* __restrict__ x, float* __restrict__ hinit) {
  const int i = blockIdx.x * blockDim.x + threadIdx.x;
  if (i >= BATCH * EMB) return;
  const int b = i >> 7, e = i & (EMB - 1);
  const float* p = &x[(size_t)b * NNODE * EMB + e];
  float s = 0.0f;
  for (int n = 0; n < NNODE; ++n) s += p[(size_t)n * EMB];
  hinit[i] = s * (1.0f / (float)NNODE);
}

// ---------------------------------------------------------------------------
// Pack a weight W (NO x NK, used as x @ W^T) into WMMA B-fragments (f16).
// B matrix is KxN with b[k,n] = W[n,k].  Dense f16 B 32x16 lane layout:
//   lanes 0-15 : n = lane,     j=0..7 -> K=kbase+j, j=8..15 -> K=kbase+16+(j-8)
//   lanes16-31 : n = lane-16,  same but K offset +8
// frag[((u*(NK/32)+t)*32 + lane)*16 + j]
// ---------------------------------------------------------------------------
__global__ void packWeightKernel(const float* __restrict__ W, _Float16* __restrict__ frag,
                                 int NO, int NK) {
  const int total = (NO / 16) * (NK / 32) * 32;
  const int i = blockIdx.x * blockDim.x + threadIdx.x;
  if (i >= total) return;
  const int lane = i & 31;
  const int ut = i >> 5;               // u*(NK/32)+t
  const int nkt = NK / 32;
  const int u = ut / nkt, t = ut % nkt;
  const int n = u * 16 + (lane & 15);
  const int kb = t * 32 + (lane >> 4) * 8;
  for (int j = 0; j < 16; ++j)
    frag[((size_t)ut * 32 + lane) * 16 + j] = (_Float16)W[(size_t)n * NK + kb + (j < 8 ? j : j + 8)];
}

__global__ void cvtF16Kernel(const float* __restrict__ src, _Float16* __restrict__ dst, int n) {
  const int i = blockIdx.x * blockDim.x + threadIdx.x;
  if (i < n) dst[i] = (_Float16)src[i];
}

// ---------------------------------------------------------------------------
// A-fragment (16x32 f16) from an LDS row-major f16 tile.
// lanes 0-15: row = lane,   elements j=0..7 -> K=kbase+j,   j=8..15 -> K=kbase+16+(j-8)
// lanes16-31: row = lane-16, same but K offset +8.
// ---------------------------------------------------------------------------
__device__ inline v16h loadAFrag(const _Float16* tile, int ldk, int kbase, int lane) {
  const int half = lane >> 4, r = lane & 15;
  const _Float16* p0 = &tile[r * ldk + kbase + half * 8];
  const _Float16* p1 = p0 + 16;
  v16h a;
#pragma unroll
  for (int j = 0; j < 8; ++j) { a[j] = p0[j]; a[8 + j] = p1[j]; }
  return a;
}

// ---------------------------------------------------------------------------
// Precompute K4/V4/Kx = x @ {Wk,Wv,WprjK}^T with WMMA f16, store f16.
// One 16-row tile per block; 8 waves cover the 128 output columns.
// A-fragment is built once per K-step and fed to three WMMAs (3x reuse).
// ---------------------------------------------------------------------------
__global__ void proj3Kernel(const float* __restrict__ x,
                            const _Float16* __restrict__ fWk,
                            const _Float16* __restrict__ fWv,
                            const _Float16* __restrict__ fWp,
                            _Float16* __restrict__ K4h,
                            _Float16* __restrict__ V4h,
                            _Float16* __restrict__ Kxh) {
  __shared__ _Float16 tile[16 * EMB];
  const size_t m0 = (size_t)blockIdx.x * 16;
  for (int idx = threadIdx.x; idx < 16 * EMB; idx += 256)
    tile[idx] = (_Float16)x[m0 * EMB + idx];
  __syncthreads();

  const int wave = threadIdx.x >> 5;       // n-tile 0..7
  const int lane = threadIdx.x & 31;
  v8f cK = {}; v8f cV = {}; v8f cP = {};
#pragma unroll
  for (int t = 0; t < 4; ++t) {
    const v16h a = loadAFrag(tile, EMB, t * 32, lane);
    const v16h bK = *(const v16h*)&fWk[(((size_t)wave * 4 + t) * 32 + lane) * 16];
    const v16h bV = *(const v16h*)&fWv[(((size_t)wave * 4 + t) * 32 + lane) * 16];
    const v16h bP = *(const v16h*)&fWp[(((size_t)wave * 4 + t) * 32 + lane) * 16];
    cK = __builtin_amdgcn_wmma_f32_16x16x32_f16(false, a, false, bK, (short)0, cK, false, false);
    cV = __builtin_amdgcn_wmma_f32_16x16x32_f16(false, a, false, bV, (short)0, cV, false, false);
    cP = __builtin_amdgcn_wmma_f32_16x16x32_f16(false, a, false, bP, (short)0, cP, false, false);
  }
  // C/D layout: element rr -> row m0+rr+(lane>=16?8:0), col wave*16+(lane&15)
  const int half = lane >> 4;
  const int col = wave * 16 + (lane & 15);
#pragma unroll
  for (int rr = 0; rr < 8; ++rr) {
    const size_t row = m0 + rr + half * 8;
    K4h[row * HD + col] = (_Float16)cK[rr];
    V4h[row * HD + col] = (_Float16)cV[rr];
    Kxh[row * HD + col] = (_Float16)cP[rr];
  }
}

// ---------------------------------------------------------------------------
// Per-step q projection: q = [hinit|last|first] @ Wq^T, M=2048 K=384 N=128.
// 16 batch rows per block, 8 waves over columns, 12 WMMA K-steps.
// ---------------------------------------------------------------------------
__global__ void qKernel(const float* __restrict__ hinit,
                        const float* __restrict__ last,
                        const float* __restrict__ first,
                        const _Float16* __restrict__ fWq,
                        _Float16* __restrict__ q_h) {
  __shared__ _Float16 tile[16 * QIN];
  const size_t b0 = (size_t)blockIdx.x * 16;
  for (int idx = threadIdx.x; idx < 16 * QIN; idx += 256) {
    const int r = idx / QIN, c = idx % QIN;
    float v;
    if (c < EMB)            v = hinit[(b0 + r) * EMB + c];
    else if (c < 2 * EMB)   v = last [(b0 + r) * EMB + (c - EMB)];
    else                    v = first[(b0 + r) * EMB + (c - 2 * EMB)];
    tile[r * QIN + c] = (_Float16)v;
  }
  __syncthreads();

  const int wave = threadIdx.x >> 5;
  const int lane = threadIdx.x & 31;
  v8f c = {};
#pragma unroll
  for (int t = 0; t < 12; ++t) {
    const v16h a = loadAFrag(tile, QIN, t * 32, lane);
    const v16h b = *(const v16h*)&fWq[(((size_t)wave * 12 + t) * 32 + lane) * 16];
    c = __builtin_amdgcn_wmma_f32_16x16x32_f16(false, a, false, b, (short)0, c, false, false);
  }
  const int half = lane >> 4;
  const int col = wave * 16 + (lane & 15);
#pragma unroll
  for (int rr = 0; rr < 8; ++rr)
    q_h[(b0 + rr + half * 8) * HD + col] = (_Float16)c[rr];
}

// ---------------------------------------------------------------------------
// Fused decode step: one wave32 per batch. Streams f16 K4/V4/Kx rows with
// b128 (K4/Kx) and b64 (V4) loads (memory-bound phase), per-head masked softmax, ctx, qv=ctx@Wo^T,
// pointer logits with tanh clip, Gumbel-argmax sampling, log-prob & state.
// Dynamic LDS: 4 batches * 2192 floats = 35072 B.
// ---------------------------------------------------------------------------
__global__ void stepKernel(int step,
                           const _Float16* __restrict__ q_h,
                           const _Float16* __restrict__ K4h,
                           const _Float16* __restrict__ V4h,
                           const _Float16* __restrict__ Kxh,
                           const _Float16* __restrict__ Woh,
                           const float* __restrict__ x,
                           unsigned char* __restrict__ mask,
                           float* __restrict__ last,
                           float* __restrict__ first,
                           float* __restrict__ logp,
                           float* __restrict__ out,
                           const int* __restrict__ clipI) {
  extern __shared__ float sm[];
  const int wave = threadIdx.x >> 5;
  const int lane = threadIdx.x & 31;
  const int b = blockIdx.x * 4 + wave;
  const size_t rowBase = (size_t)b * NNODE;

  float* Ua = sm + wave * 2192;       // [8][200] U -> att
  float* lg = Ua + 1600;              // [200] logits
  float* qs = lg + 200;               // [128]
  float* qv = qs + 128;               // [128]
  float* cx = qv + 128;               // [128]
  const float clipf = (float)clipI[0];

  // phase 0: q to LDS (f32)
#pragma unroll
  for (int j = 0; j < 4; ++j)
    qs[lane * 4 + j] = (float)q_h[(size_t)b * HD + lane * 4 + j];
  __syncthreads();

  // phase 1: U[h,n] = (q_h . K4[b,n,h,:]) / sqrt(D), masked
  for (int it = 0; it < 7; ++it) {
    const int n = lane + it * 32;
    if (n < NNODE) {
      const float4* kr = (const float4*)&K4h[(rowBase + n) * HD];
      float acc[HEADS];
#pragma unroll
      for (int h = 0; h < HEADS; ++h) acc[h] = 0.0f;
#pragma unroll
      for (int c = 0; c < 16; ++c) {
        F4H8 t; t.v = kr[c];
        const int h = c >> 1, dbase = (c & 1) * 8;
#pragma unroll
        for (int j = 0; j < 8; ++j) acc[h] += qs[h * 16 + dbase + j] * (float)t.h[j];
      }
      const bool mk = mask[rowBase + n] != 0;
#pragma unroll
      for (int h = 0; h < HEADS; ++h)
        Ua[h * NNODE + n] = mk ? -1e30f : acc[h] * 0.25f;   // 1/sqrt(16)
    }
  }
  __syncthreads();

  // phase 2: per-head softmax, in place (U -> att)
#pragma unroll
  for (int h = 0; h < HEADS; ++h) {
    float m = -1e30f;
    for (int n = lane; n < NNODE; n += 32) m = fmaxf(m, Ua[h * NNODE + n]);
    for (int o = 16; o; o >>= 1) m = fmaxf(m, __shfl_xor(m, o, 32));
    float s = 0.0f;
    for (int n = lane; n < NNODE; n += 32) s += expf(Ua[h * NNODE + n] - m);
    for (int o = 16; o; o >>= 1) s += __shfl_xor(s, o, 32);
    const float inv = 1.0f / s;
    for (int n = lane; n < NNODE; n += 32)
      Ua[h * NNODE + n] = expf(Ua[h * NNODE + n] - m) * inv;
  }
  __syncthreads();

  // phase 3: ctx = att . V4 — lane owns 4 contiguous components (coalesced 8B/row)
  {
    const int h = lane >> 2;                    // 4 comps per lane stay in one head
    const float* attRow = &Ua[h * NNODE];
    float c0 = 0, c1 = 0, c2 = 0, c3 = 0;
    for (int n = 0; n < NNODE; ++n) {
      const float a = attRow[n];
      union { float2 v; _Float16 h4[4]; } vv;
      vv.v = *(const float2*)&V4h[(rowBase + n) * HD + lane * 4];
      c0 += a * (float)vv.h4[0]; c1 += a * (float)vv.h4[1];
      c2 += a * (float)vv.h4[2]; c3 += a * (float)vv.h4[3];
    }
    cx[lane * 4 + 0] = c0; cx[lane * 4 + 1] = c1;
    cx[lane * 4 + 2] = c2; cx[lane * 4 + 3] = c3;
  }
  __syncthreads();

  // phase 4: qv[e] = sum_j ctx[j] * Wo[e,j]  (Wo f16 stays hot in L2/WGP$)
#pragma unroll
  for (int eo = 0; eo < 4; ++eo) {
    const int e = lane + eo * 32;
    const float4* wr = (const float4*)&Woh[(size_t)e * HD];
    float s = 0.0f;
#pragma unroll
    for (int c = 0; c < 16; ++c) {
      F4H8 t; t.v = wr[c];
#pragma unroll
      for (int j = 0; j < 8; ++j) s += cx[c * 8 + j] * (float)t.h[j];
    }
    qv[e] = s;
  }
  __syncthreads();

  // phase 5: logits[n] = clip * tanh((qv . Kx[b,n]) / E), masked
  for (int it = 0; it < 7; ++it) {
    const int n = lane + it * 32;
    if (n < NNODE) {
      const float4* xr = (const float4*)&Kxh[(rowBase + n) * HD];
      float s = 0.0f;
#pragma unroll
      for (int c = 0; c < 16; ++c) {
        F4H8 t; t.v = xr[c];
#pragma unroll
        for (int j = 0; j < 8; ++j) s += qv[c * 8 + j] * (float)t.h[j];
      }
      const float uu = clipf * tanhf(s * (1.0f / (float)EMB));
      lg[n] = mask[rowBase + n] ? -1e30f : uu;
    }
  }
  __syncthreads();

  // phase 6: Gumbel-argmax sampling + log-softmax accumulation
  U2 k0; k0.x = 0u; k0.y = 42u;
  U2 ci; ci.x = 0u; ci.y = (unsigned)step;
  const U2 kk = tf2x32(k0, ci);                 // per-step folded key
  float bestV = -3.0e38f; int bestI = 0;
  float mx = -1e30f;
  for (int it = 0; it < 7; ++it) {
    const int n = lane + it * 32;
    if (n < NNODE) {
      const float l = lg[n];
      mx = fmaxf(mx, l);
      U2 ctr; ctr.x = (unsigned)b; ctr.y = (unsigned)n;
      const U2 rb = tf2x32(kk, ctr);
      const float u01 = (float)(rb.x >> 8) * (1.0f / 16777216.0f);
      const float g = -logf(-logf(fmaxf(u01, 1e-12f)));
      const float v = l + g;
      if (v > bestV || (v == bestV && n < bestI)) { bestV = v; bestI = n; }
    }
  }
  for (int o = 16; o; o >>= 1) {
    const float ov = __shfl_xor(bestV, o, 32);
    const int   oi = __shfl_xor(bestI, o, 32);
    if (ov > bestV || (ov == bestV && oi < bestI)) { bestV = ov; bestI = oi; }
    mx = fmaxf(mx, __shfl_xor(mx, o, 32));
  }
  float se = 0.0f;
  for (int n = lane; n < NNODE; n += 32) se += expf(lg[n] - mx);
  for (int o = 16; o; o >>= 1) se += __shfl_xor(se, o, 32);

  const int nxt = bestI;
  if (lane == 0) {
    const float lp = lg[nxt] - mx - logf(se);
    const float acc = (step == 0 ? 0.0f : logp[b]) + lp;
    logp[b] = acc;
    out[(size_t)b * NNODE + step] = (float)nxt;           // visited.T[b, i]
    if (step == NNODE - 1) out[(size_t)BATCH * NNODE + b] = acc;  // logp[b]
    mask[rowBase + nxt] = 1;
  }
  // state update: last = x[b,nxt]; first = last if step==0
  const float* xr = &x[(rowBase + (size_t)nxt) * EMB];
#pragma unroll
  for (int j = 0; j < 4; ++j) {
    const float v = xr[lane * 4 + j];
    last[(size_t)b * EMB + lane * 4 + j] = v;
    if (step == 0) first[(size_t)b * EMB + lane * 4 + j] = v;
  }
}

// ---------------------------------------------------------------------------
extern "C" void kernel_launch(void* const* d_in, const int* in_sizes, int n_in,
                              void* d_out, int out_size, void* d_ws, size_t ws_size,
                              hipStream_t stream) {
  const float* x     = (const float*)d_in[0];
  const float* initK = (const float*)d_in[1];
  const float* initV = (const float*)d_in[2];
  const float* WprjK = (const float*)d_in[3];
  const float* Wq    = (const float*)d_in[4];
  const float* Wk    = (const float*)d_in[5];
  const float* Wv    = (const float*)d_in[6];
  const float* Wo    = (const float*)d_in[7];
  const int*   clip  = (const int*)d_in[8];
  float* out = (float*)d_out;

  // workspace bump allocator (256B aligned)
  uintptr_t p = (uintptr_t)d_ws;
  auto alloc = [&](size_t bytes) -> void* {
    p = (p + 255) & ~(uintptr_t)255;
    void* r = (void*)p; p += bytes; return r;
  };
  const size_t MR = (size_t)BATCH * NNODE;      // 409600 rows
  _Float16* fWk  = (_Float16*)alloc(8 * 4 * 32 * 16 * sizeof(_Float16));
  _Float16* fWv  = (_Float16*)alloc(8 * 4 * 32 * 16 * sizeof(_Float16));
  _Float16* fWp  = (_Float16*)alloc(8 * 4 * 32 * 16 * sizeof(_Float16));
  _Float16* fWq  = (_Float16*)alloc(8 * 12 * 32 * 16 * sizeof(_Float16));
  _Float16* Woh  = (_Float16*)alloc((size_t)EMB * HD * sizeof(_Float16));
  float*    hinit= (float*)alloc((size_t)BATCH * EMB * sizeof(float));
  float*    last = (float*)alloc((size_t)BATCH * EMB * sizeof(float));
  float*    first= (float*)alloc((size_t)BATCH * EMB * sizeof(float));
  _Float16* q_h  = (_Float16*)alloc((size_t)BATCH * HD * sizeof(_Float16));
  _Float16* K4h  = (_Float16*)alloc(MR * HD * sizeof(_Float16));
  _Float16* V4h  = (_Float16*)alloc(MR * HD * sizeof(_Float16));
  _Float16* Kxh  = (_Float16*)alloc(MR * HD * sizeof(_Float16));
  unsigned char* mask = (unsigned char*)alloc(MR);
  float*    logp = (float*)alloc((size_t)BATCH * sizeof(float));
  if (p - (uintptr_t)d_ws > ws_size) return;    // scratch too small: bail deterministically

  // one-time (per launch) precompute
  initStateKernel<<<(BATCH * NNODE + 255) / 256, 256, 0, stream>>>(initK, initV, last, first, mask, logp);
  hinitKernel<<<(BATCH * EMB + 255) / 256, 256, 0, stream>>>(x, hinit);
  packWeightKernel<<<(8 * 4 * 32 + 255) / 256, 256, 0, stream>>>(Wk, fWk, HD, EMB);
  packWeightKernel<<<(8 * 4 * 32 + 255) / 256, 256, 0, stream>>>(Wv, fWv, HD, EMB);
  packWeightKernel<<<(8 * 4 * 32 + 255) / 256, 256, 0, stream>>>(WprjK, fWp, EMB, EMB);
  packWeightKernel<<<(8 * 12 * 32 + 255) / 256, 256, 0, stream>>>(Wq, fWq, HD, QIN);
  cvtF16Kernel<<<(EMB * HD + 255) / 256, 256, 0, stream>>>(Wo, Woh, EMB * HD);
  proj3Kernel<<<(int)(MR / 16), 256, 0, stream>>>(x, fWk, fWv, fWp, K4h, V4h, Kxh);

  // sequential decode: 200 steps, 2 kernels/step, all stream-ordered
  const size_t stepShmem = 4 * 2192 * sizeof(float);   // 35072 B
  for (int i = 0; i < NNODE; ++i) {
    qKernel<<<BATCH / 16, 256, 0, stream>>>(hinit, last, first, fWq, q_h);
    stepKernel<<<BATCH / 4, 128, stepShmem, stream>>>(i, q_h, K4h, V4h, Kxh, Woh, x,
                                                      mask, last, first, logp, out, clip);
  }
}
